// InundationBlock_50972671869438
// MI455X (gfx1250) — matrix operand order repaired and checked by reading, not compiled, not run on hardware
//
#include <hip/hip_runtime.h>
#include <hip/hip_bf16.h>
#include <math.h>

typedef __attribute__((ext_vector_type(16))) _Float16 v16h;
typedef __attribute__((ext_vector_type(8)))  _Float16 v8h;
typedef __attribute__((ext_vector_type(8)))  float    v8f;

#define NN_ 10000
#define NE_ 160000
#define TT_ 64
#define FIN_ 16
#define HH_ 128

#define TILE_M 128   // 8 waves x 16-row strip
#define TILE_N 128

// ---------------------------------------------------------------------------
// LDS-free WMMA GEMM, compile-time K/N: C[M,N] = A[M,K](f16, stride lda) * Bp
// (f16, pre-packed WMMA fragment order). 256 threads = 8 wave32; block tile
// 128x128, wave tile 16x128 (8 independent acc chains -> no WMMA hazard NOPs;
// one A fragment feeds 8 WMMAs). Full K-unroll -> every load/store is a single
// base + compile-time immediate offset.
//
// Packed-B layout: fragment (kt, jt) for lane L is 16 contiguous f16 at
//   Bp + ((kt*NT + jt)*32 + L)*16      (L = (n&15)+16*((k>>3)&1), p = (k&7)+8*((k>>4)&1))
// A fragment: lane (m=L&15, hi=L>>4): slots 0..7 = A[m][k0+8hi .. +7],
//   slots 8..15 = A[m][k0+16+8hi .. +7]  -> two 16B loads per lane.
// ---------------------------------------------------------------------------
template <int KK, int NCOLS>
__global__ void __launch_bounds__(256)
gemm_wmma(const _Float16* __restrict__ A, int lda, int M,
          const _Float16* __restrict__ Bp, float* __restrict__ C)
{
    constexpr int NT = NCOLS / 16;
    constexpr int KT = (KK + 31) / 32;

    const int tid  = threadIdx.x;
    const int lane = tid & 31;
    const int wav  = tid >> 5;          // 0..7 : 16-row strip
    const int row0 = blockIdx.y * TILE_M;
    const int col0 = blockIdx.x * TILE_N;
    const int ml   = lane & 15;
    const int hi   = lane >> 4;

    int m = row0 + wav * 16 + ml;
    if (m >= M) m = M - 1;              // clamp: rows >= M never stored
    const _Float16* __restrict__ ar = A + (long)m * lda + 8 * hi;
    const _Float16* __restrict__ bb = Bp + ((long)(col0 >> 4) * 32 + lane) * 16;

    v8f acc[8];
#pragma unroll
    for (int j = 0; j < 8; ++j)
#pragma unroll
        for (int r = 0; r < 8; ++r) acc[j][r] = 0.0f;

#pragma unroll
    for (int kt = 0; kt < KT; ++kt) {
        union { v16h v; v8h h[2]; } ua;
        ua.h[0] = *(const v8h*)(ar + kt * 32);        // slots 0..7
        ua.h[1] = *(const v8h*)(ar + kt * 32 + 16);   // slots 8..15
#pragma unroll
        for (int j = 0; j < 8; ++j) {
            const v16h bf = *(const v16h*)(bb + (kt * NT + j) * 512);
            acc[j] = __builtin_amdgcn_wmma_f32_16x16x32_f16(
                false, ua.v, false, bf, (short)0, acc[j], false, false);
        }
    }

    // C/D layout: VGPR r -> M = r + 8*hi, N = lane&15.
    // Single base pointer; all offsets compile-time immediates.
    float* __restrict__ cp =
        C + (long)(row0 + wav * 16 + 8 * hi) * NCOLS + col0 + ml;
    if (row0 + TILE_M <= M) {            // block-uniform fast path (no guards)
#pragma unroll
        for (int j = 0; j < 8; ++j)
#pragma unroll
            for (int r = 0; r < 8; ++r)
                cp[j * 16 + r * NCOLS] = acc[j][r];
    } else {
        const int rbase = row0 + wav * 16 + 8 * hi;
#pragma unroll
        for (int j = 0; j < 8; ++j)
#pragma unroll
            for (int r = 0; r < 8; ++r)
                if (rbase + r < M) cp[j * 16 + r * NCOLS] = acc[j][r];
    }
}

// ---------------------------------------------------------------------------
// Pack rows k in [k0, k0+Kr) of the fragment-ordered B buffer (N cols, NT=N/16
// n-tiles). Source f32: trans ? src[n*lds + kl] : src[kl*N + n]; kl >= Kvalid
// writes zeros (K-tail padding).
// ---------------------------------------------------------------------------
__global__ void k_pack_b(_Float16* __restrict__ dst, const float* __restrict__ src,
                         int N, int NT, int k0, int Kr, int Kvalid, int lds, int trans) {
    int i = blockIdx.x * blockDim.x + threadIdx.x;
    if (i >= Kr * N) return;
    int kl = i / N, n = i % N;
    int k  = k0 + kl;
    float v = 0.0f;
    if (kl < Kvalid)
        v = trans ? src[(long)n * lds + kl] : src[(long)kl * N + n];
    int kt = k >> 5, jt = n >> 4;
    int L  = (n & 15) + 16 * ((k >> 3) & 1);
    int p  = (k & 7) + 8 * ((k >> 4) & 1);
    dst[((long)(kt * NT + jt) * 32 + L) * 16 + p] = (_Float16)v;
}

// ---------------------------------------------------------------------------
// Helper kernels
// ---------------------------------------------------------------------------
__global__ void k_fill(float* p, float v, int n) {
    int i = blockIdx.x * blockDim.x + threadIdx.x;
    if (i < n) p[i] = v;
}

__global__ void k_deg(const int* __restrict__ dst, float* deg, int E) {
    int e = blockIdx.x * blockDim.x + threadIdx.x;
    if (e < E) atomicAdd(&deg[dst[e]], 1.0f);
}

__global__ void k_rsqrt(float* dinv, const float* __restrict__ deg, int n) {
    int i = blockIdx.x * blockDim.x + threadIdx.x;
    if (i < n) dinv[i] = rsqrtf(deg[i]);   // deg >= 1 (self loop)
}

__global__ void k_cvt(_Float16* d, const float* __restrict__ s, int n) {
    int i = blockIdx.x * blockDim.x + threadIdx.x;
    if (i < n) d[i] = (_Float16)s[i];
}

// agg[n,f] = h[n,f] * dinv[n]^2   (self-loop term; initializes agg)
__global__ void k_self(float* agg, const float* __restrict__ h,
                       const float* __restrict__ dinv, int n) {
    int i = blockIdx.x * blockDim.x + threadIdx.x;
    if (i < n) {
        float dv = dinv[i >> 7];
        agg[i] = h[i] * dv * dv;
    }
}

// edge-parallel scatter: agg[dst] += h[src] * dinv[src]*dinv[dst]; 2 edges/block
__global__ void k_scatter(const int* __restrict__ src, const int* __restrict__ dst,
                          const float* __restrict__ dinv, const float* __restrict__ h,
                          float* agg, int E) {
    int e = blockIdx.x * 2 + (threadIdx.x >> 7);
    int f = threadIdx.x & 127;
    if (e < E) {
        int s = src[e], d = dst[e];
        float w = dinv[s] * dinv[d];
        atomicAdd(&agg[(long)d * HH_ + f], h[(long)s * HH_ + f] * w);
    }
}

// h1h = f16(relu(agg + b1))  (feeds next GEMM only -> f16)
__global__ void k_bias_relu(const float* __restrict__ agg, const float* __restrict__ b,
                            _Float16* h1h, int n) {
    int i = blockIdx.x * blockDim.x + threadIdx.x;
    if (i < n) h1h[i] = (_Float16)fmaxf(agg[i] + b[i & 127], 0.0f);
}

// g_t = layernorm(agg + b2, ln2) -> f16 into cat[:, 0:128]; one node per block
__global__ void k_bias_ln_cat(const float* __restrict__ agg, const float* __restrict__ b,
                              const float* __restrict__ lg, const float* __restrict__ lb,
                              _Float16* cat) {
    __shared__ float s1[128], s2[128];
    int n = blockIdx.x, f = threadIdx.x;
    float v = agg[(long)n * HH_ + f] + b[f];
    s1[f] = v; s2[f] = v * v;
    __syncthreads();
    for (int st = 64; st > 0; st >>= 1) {
        if (f < st) { s1[f] += s1[f + st]; s2[f] += s2[f + st]; }
        __syncthreads();
    }
    float mean = s1[0] * (1.0f / 128.0f);
    float var  = s2[0] * (1.0f / 128.0f) - mean * mean;
    float o = (v - mean) * rsqrtf(var + 1e-5f) * lg[f] + lb[f];
    cat[(long)n * 256 + f] = (_Float16)o;
}

__global__ void k_init_state(float* hbuf, float* cbuf, _Float16* cat, int n) {
    int i = blockIdx.x * blockDim.x + threadIdx.x;
    if (i < n) {
        hbuf[i] = 0.0f; cbuf[i] = 0.0f;
        cat[(long)(i >> 7) * 256 + 128 + (i & 127)] = (_Float16)0.0f;
    }
}

__device__ __forceinline__ float sigm(float x) { return 1.0f / (1.0f + expf(-x)); }

// LSTM gates + state update + ln1 layernorm of h -> series; one node per block
__global__ void k_lstm(const float* __restrict__ zc, const float* __restrict__ bih,
                       const float* __restrict__ bhh, float* hbuf, float* cbuf,
                       _Float16* cat, const float* __restrict__ lg,
                       const float* __restrict__ lb, float* series, int t) {
    __shared__ float s1[128], s2[128];
    int n = blockIdx.x, f = threadIdx.x;
    const float* z = zc + (long)n * 512;
    float ig = sigm(z[f]         + bih[f]         + bhh[f]);
    float fg = sigm(z[128 + f]   + bih[128 + f]   + bhh[128 + f]);
    float gg = tanhf(z[256 + f]  + bih[256 + f]   + bhh[256 + f]);
    float og = sigm(z[384 + f]   + bih[384 + f]   + bhh[384 + f]);
    float c  = fg * cbuf[(long)n * HH_ + f] + ig * gg;
    float h  = og * tanhf(c);
    cbuf[(long)n * HH_ + f] = c;
    hbuf[(long)n * HH_ + f] = h;
    cat[(long)n * 256 + 128 + f] = (_Float16)h;   // next-step GEMM input

    s1[f] = h; s2[f] = h * h;
    __syncthreads();
    for (int st = 64; st > 0; st >>= 1) {
        if (f < st) { s1[f] += s1[f + st]; s2[f] += s2[f + st]; }
        __syncthreads();
    }
    float mean = s1[0] * (1.0f / 128.0f);
    float var  = s2[0] * (1.0f / 128.0f) - mean * mean;
    series[(long)n * TT_ * HH_ + (long)t * HH_ + f] =
        (h - mean) * rsqrtf(var + 1e-5f) * lg[f] + lb[f];
}

__global__ void k_head(const float* __restrict__ mm, const float* __restrict__ b,
                       float* out, int n, int do_tanh) {
    int i = blockIdx.x * blockDim.x + threadIdx.x;
    if (i < n) {
        float v = mm[i] + b[i & 127];
        out[i] = do_tanh ? tanhf(v) : v;
    }
}

// ---------------------------------------------------------------------------
extern "C" void kernel_launch(void* const* d_in, const int* in_sizes, int n_in,
                              void* d_out, int out_size, void* d_ws, size_t ws_size,
                              hipStream_t stream) {
    (void)in_sizes; (void)n_in; (void)out_size; (void)ws_size;
    const float* inputs = (const float*)d_in[0];
    const int*   edges  = (const int*)d_in[1];
    const int*   esrc   = edges;
    const int*   edst   = edges + NE_;
    const float* W1 = (const float*)d_in[2];
    const float* b1 = (const float*)d_in[3];
    const float* W2 = (const float*)d_in[4];
    const float* b2 = (const float*)d_in[5];
    const float* Wih = (const float*)d_in[6];
    const float* Whh = (const float*)d_in[7];
    const float* bih = (const float*)d_in[8];
    const float* bhh = (const float*)d_in[9];
    const float* ln1g = (const float*)d_in[10];
    const float* ln1b = (const float*)d_in[11];
    const float* ln2g = (const float*)d_in[12];
    const float* ln2b = (const float*)d_in[13];
    const float* Wh = (const float*)d_in[14];
    const float* bh = (const float*)d_in[15];
    const float* Wc = (const float*)d_in[16];
    const float* bc = (const float*)d_in[17];

    float* out    = (float*)d_out;
    float* series = out;                                    // [N,T,H]
    float* hidden = out + (size_t)NN_ * TT_ * HH_;          // [1,N,H]
    float* cell   = hidden + (size_t)NN_ * HH_;             // [1,N,H]

    char* ws = (char*)d_ws;
    size_t off = 0;
    auto carve = [&](size_t bytes) -> char* {
        char* p = ws + off;
        off = (off + bytes + 255) & ~(size_t)255;
        return p;
    };
    float*    deg   = (float*)carve((size_t)NN_ * 4);
    float*    dinv  = (float*)carve((size_t)NN_ * 4);
    float*    mm    = (float*)carve((size_t)NN_ * HH_ * 4);        // GEMM out scratch
    float*    agg   = (float*)carve((size_t)NN_ * HH_ * 4);        // scatter accum
    float*    zc    = (float*)carve((size_t)NN_ * 4 * HH_ * 4);    // [N,4H]
    float*    hbuf  = (float*)carve((size_t)NN_ * HH_ * 4);
    float*    cbuf  = (float*)carve((size_t)NN_ * HH_ * 4);
    _Float16* xin_h = (_Float16*)carve((size_t)NN_ * TT_ * FIN_ * 2); // f16 inputs
    _Float16* h1h   = (_Float16*)carve((size_t)NN_ * HH_ * 2);
    _Float16* cat   = (_Float16*)carve((size_t)NN_ * 2 * HH_ * 2);   // [N,2H] f16
    _Float16* hTh   = (_Float16*)carve((size_t)NN_ * HH_ * 2);
    _Float16* cTh   = (_Float16*)carve((size_t)NN_ * HH_ * 2);
    _Float16* W1p   = (_Float16*)carve((size_t)32 * HH_ * 2);        // K padded to 32
    _Float16* W2p   = (_Float16*)carve((size_t)HH_ * HH_ * 2);
    _Float16* Wcatp = (_Float16*)carve((size_t)2 * HH_ * 4 * HH_ * 2); // [256,512]
    _Float16* Whp   = (_Float16*)carve((size_t)HH_ * HH_ * 2);
    _Float16* Wcp   = (_Float16*)carve((size_t)HH_ * HH_ * 2);

    const int TPB = 256;
    auto cdiv = [](int a, int b) { return (a + b - 1) / b; };
    const int NH = NN_ * HH_;

    // --- prep: GCN norm, f16 input staging, fragment-packed weights ---
    k_fill<<<cdiv(NN_, TPB), TPB, 0, stream>>>(deg, 1.0f, NN_);     // self loops
    k_deg<<<cdiv(NE_, TPB), TPB, 0, stream>>>(edst, deg, NE_);
    k_rsqrt<<<cdiv(NN_, TPB), TPB, 0, stream>>>(dinv, deg, NN_);
    k_cvt<<<cdiv(NN_ * TT_ * FIN_, TPB), TPB, 0, stream>>>(xin_h, inputs, NN_ * TT_ * FIN_);
    // W1 [16,128] no-trans, K padded to 32
    k_pack_b<<<cdiv(32 * HH_, TPB), TPB, 0, stream>>>(W1p, W1, HH_, 8, 0, 32, FIN_, 0, 0);
    // W2 [128,128] no-trans
    k_pack_b<<<cdiv(HH_ * HH_, TPB), TPB, 0, stream>>>(W2p, W2, HH_, 8, 0, HH_, HH_, 0, 0);
    // Wcat [256,512] = [Wih^T ; Whh^T]  (sources [512,128], transposed)
    k_pack_b<<<cdiv(HH_ * 4 * HH_, TPB), TPB, 0, stream>>>(Wcatp, Wih, 4 * HH_, 32, 0,   HH_, HH_, HH_, 1);
    k_pack_b<<<cdiv(HH_ * 4 * HH_, TPB), TPB, 0, stream>>>(Wcatp, Whh, 4 * HH_, 32, HH_, HH_, HH_, HH_, 1);
    // Wh^T, Wc^T [128,128]
    k_pack_b<<<cdiv(HH_ * HH_, TPB), TPB, 0, stream>>>(Whp, Wh, HH_, 8, 0, HH_, HH_, HH_, 1);
    k_pack_b<<<cdiv(HH_ * HH_, TPB), TPB, 0, stream>>>(Wcp, Wc, HH_, 8, 0, HH_, HH_, HH_, 1);
    k_init_state<<<cdiv(NH, TPB), TPB, 0, stream>>>(hbuf, cbuf, cat, NH);

    dim3 g1(1, cdiv(NN_, TILE_M));   // 128 output cols
    dim3 g4(4, cdiv(NN_, TILE_M));   // 512 output cols

    for (int t = 0; t < TT_; ++t) {
        // conv1: x_t @ W1 (rows strided through f16 [N,T,16]; K=16 zero-padded)
        gemm_wmma<32, 128><<<g1, 256, 0, stream>>>(xin_h + (size_t)t * FIN_,
                                                   TT_ * FIN_, NN_, W1p, mm);
        k_self<<<cdiv(NH, TPB), TPB, 0, stream>>>(agg, mm, dinv, NH);
        k_scatter<<<cdiv(NE_, 2), 256, 0, stream>>>(esrc, edst, dinv, mm, agg, NE_);
        k_bias_relu<<<cdiv(NH, TPB), TPB, 0, stream>>>(agg, b1, h1h, NH);
        // conv2: h1 @ W2
        gemm_wmma<128, 128><<<g1, 256, 0, stream>>>(h1h, HH_, NN_, W2p, mm);
        k_self<<<cdiv(NH, TPB), TPB, 0, stream>>>(agg, mm, dinv, NH);
        k_scatter<<<cdiv(NE_, 2), 256, 0, stream>>>(esrc, edst, dinv, mm, agg, NE_);
        // + b2, ln2 layernorm -> f16 cat[:,0:128]
        k_bias_ln_cat<<<NN_, 128, 0, stream>>>(agg, b2, ln2g, ln2b, cat);
        // fused LSTM GEMM: [g_t | h] [N,256] @ [Wih^T; Whh^T] [256,512]
        gemm_wmma<256, 512><<<g4, 256, 0, stream>>>(cat, 2 * HH_, NN_, Wcatp, zc);
        k_lstm<<<NN_, 128, 0, stream>>>(zc, bih, bhh, hbuf, cbuf, cat,
                                        ln1g, ln1b, series, t);
    }

    // heads: hidden = tanh(hT @ Wh^T + bh), cell = cT @ Wc^T + bc
    k_cvt<<<cdiv(NH, TPB), TPB, 0, stream>>>(hTh, hbuf, NH);
    k_cvt<<<cdiv(NH, TPB), TPB, 0, stream>>>(cTh, cbuf, NH);
    gemm_wmma<128, 128><<<g1, 256, 0, stream>>>(hTh, HH_, NN_, Whp, mm);
    k_head<<<cdiv(NH, TPB), TPB, 0, stream>>>(mm, bh, hidden, NH, 1);
    gemm_wmma<128, 128><<<g1, 256, 0, stream>>>(cTh, HH_, NN_, Wcp, mm);
    k_head<<<cdiv(NH, TPB), TPB, 0, stream>>>(mm, bc, cell, NH, 0);
}
